// AttentionModel_13288628814038
// MI455X (gfx1250) — matrix operand order repaired and checked by reading, not compiled, not run on hardware
//
#include <hip/hip_runtime.h>
#include <math.h>
#include <stddef.h>

// ---------------- model constants ----------------
constexpr int Bsz  = 16;
constexpr int Sseq = 512;
constexpr int FC   = 24;
constexpr int FO   = 16;
constexpr int Dm   = 512;
constexpr int Hh   = 8;
constexpr int DKh  = 64;
constexpr int FFd  = 2048;
constexpr int Lnum = 4;
constexpr int HD   = Hh * DKh;     // 512
constexpr float EPSf = 1e-6f;

// ---------------- types ----------------
typedef __bf16 bf16;
typedef bf16  v16bf __attribute__((ext_vector_type(16)));
typedef float v8f   __attribute__((ext_vector_type(8)));
typedef unsigned int u32x4 __attribute__((ext_vector_type(4)));
typedef int i32x4 __attribute__((ext_vector_type(4)));
typedef int i32x8 __attribute__((ext_vector_type(8)));

static __device__ __forceinline__ v8f wmma_bf16(v16bf a, v16bf b, v8f c) {
  return __builtin_amdgcn_wmma_f32_16x16x32_bf16(
      false, a, false, b, (short)0, c, false, false);
}

// A fragment (16x32 bf16, row-major source). lane = m + 16*g (ISA 7.12.2):
// halves 0..7 -> K = 8*g + i ; halves 8..15 -> K = 16 + 8*g + i
static __device__ __forceinline__ v16bf load_a_frag_bf16(const bf16* __restrict__ row, int g) {
  v16bf f;
#pragma unroll
  for (int i = 0; i < 8; ++i) {
    f[i]     = row[8 * g + i];
    f[i + 8] = row[16 + 8 * g + i];
  }
  return f;
}

// A fragment from fp32 source (used for softmax probs held in LDS as f32)
static __device__ __forceinline__ v16bf load_a_frag_f32(const float* __restrict__ row, int g) {
  v16bf f;
#pragma unroll
  for (int i = 0; i < 8; ++i) {
    f[i]     = (bf16)row[8 * g + i];
    f[i + 8] = (bf16)row[16 + 8 * g + i];
  }
  return f;
}

// B fragment (32x16, LDS row stride 64): lane = n + 16*g; halves 0..7 -> K = 8*g + i ; halves 8..15 -> K = 16 + 8*g + i
static __device__ __forceinline__ v16bf load_b_frag_bf16(const bf16* __restrict__ p, int g) {
  v16bf f;
#pragma unroll
  for (int i = 0; i < 8; ++i) {
    f[i]     = p[(8 * g + i) * 64];
    f[i + 8] = p[(16 + 8 * g + i) * 64];
  }
  return f;
}

// ---------------- Tensor Data Mover: stage a 32(K) x 64(N) bf16 tile into LDS ----------------
// D# per CDNA5 ISA: group0 = {count/lds_addr/global_addr/type},
// group1 = {data_size=2B, tensor_dim0=N, tensor_dim1=K, tile 64x32, stride0=N}.
static __device__ __forceinline__ void tdm_load_b_tile(unsigned lds_off,
                                                       const bf16* gptr,
                                                       int N, int K) {
  unsigned long long ga = (unsigned long long)(size_t)gptr;
  u32x4 g0;
  g0[0] = 1u;                                        // count=1, is_restore=0
  g0[1] = lds_off;                                   // lds_addr (bytes)
  g0[2] = (unsigned)ga;                              // global_addr[31:0]
  g0[3] = (unsigned)((ga >> 32) & 0x01FFFFFFu)       // global_addr[56:32]
        | 0x80000000u;                               // type=2 ("image")
  i32x8 g1;
  g1[0] = 0x10000;                                   // data_size=1 (2 bytes), mask=0
  g1[1] = (N & 0xFFFF) << 16;                        // tensor_dim0[15:0]
  g1[2] = ((unsigned)N >> 16) | ((K & 0xFFFF) << 16);// tensor_dim0[31:16] | tensor_dim1[15:0]
  g1[3] = ((unsigned)K >> 16) | (64u << 16);         // tensor_dim1[31:16] | tile_dim0=64
  g1[4] = 32;                                        // tile_dim1=32, tile_dim2=0
  g1[5] = N;                                         // tensor_dim0_stride[31:0]
  g1[6] = 0;                                         // stride0 hi | stride1 lo
  g1[7] = 0;
  i32x4 gz = {0, 0, 0, 0};
#if defined(__clang_major__) && (__clang_major__ >= 23)
  i32x8 gz8 = {0, 0, 0, 0, 0, 0, 0, 0};
  __builtin_amdgcn_tensor_load_to_lds(g0, g1, gz, gz, gz8, 0);
#else
  __builtin_amdgcn_tensor_load_to_lds(g0, g1, gz, gz, 0);
#endif
}

// ---------------- fp32 -> bf16 conversion ----------------
__global__ void cvt_bf16_kernel(const float* __restrict__ in, bf16* __restrict__ out, int n) {
  int i = blockIdx.x * blockDim.x + threadIdx.x;
  if (i < n) out[i] = (bf16)in[i];
}

// ---------------- embedding: x = x_cgm @ cgm_W + cgm_b (fp32 + bf16 outputs) ----------------
__global__ void embed_kernel(const float* __restrict__ xc, const float* __restrict__ W,
                             const float* __restrict__ b, float* __restrict__ out,
                             bf16* __restrict__ outb) {
  int idx = blockIdx.x * blockDim.x + threadIdx.x;   // row*512 + d
  int row = idx >> 9, d = idx & 511;
  const float* xr = xc + (size_t)row * FC;
  float acc = b[d];
#pragma unroll
  for (int k = 0; k < FC; ++k) acc += xr[k] * W[k * Dm + d];
  out[idx]  = acc;
  outb[idx] = (bf16)acc;
}

// ---------------- pos_bias[i,j] = mean_d rel_emb[i-j+511, d] ----------------
__global__ void posbias_kernel(const float* __restrict__ rel, float* __restrict__ pb) {
  int idx = blockIdx.x * blockDim.x + threadIdx.x;
  int i = idx >> 9, j = idx & 511;
  const float* r = rel + (size_t)(i - j + Sseq - 1) * DKh;
  float s = 0.f;
#pragma unroll
  for (int t = 0; t < DKh; ++t) s += r[t];
  pb[idx] = s * (1.0f / DKh);
}

// ---------------- WMMA GEMM, bf16 operands, TDM-staged B tiles ----------------
// block: 128 threads = 4 waves; wave w -> rows [bx*64 + w*16), cols [by*64, +64)
// B tile (32x64 bf16 = 4KB) double-buffered in LDS via tensor_load_to_lds.
__global__ void __launch_bounds__(128)
gemm_wmma_bf16(const bf16* __restrict__ A, const bf16* __restrict__ Bw,
               const float* __restrict__ bias, float* __restrict__ Cf,
               bf16* __restrict__ Cb, int M, int N, int K) {
  __shared__ __align__(16) bf16 Bt[2][32][64];
  int wave = threadIdx.x >> 5, lane = threadIdx.x & 31;
  int m0 = blockIdx.x * 64 + wave * 16;
  int n0 = blockIdx.y * 64;
  int mrow = lane & 15, g = lane >> 4;
  v8f acc[4] = {};
  const bf16* Arow = A + (size_t)(m0 + mrow) * K;
  const int nk = K >> 5;
  if (wave == 0) {
    tdm_load_b_tile((unsigned)(size_t)&Bt[0][0][0], Bw + n0, N, K);
    __builtin_amdgcn_s_wait_tensorcnt(0);
  }
  __syncthreads();
  for (int kt = 0; kt < nk; ++kt) {
    int cur = kt & 1;
    if (wave == 0 && kt + 1 < nk)
      tdm_load_b_tile((unsigned)(size_t)&Bt[cur ^ 1][0][0],
                      Bw + (size_t)(kt + 1) * 32 * N + n0, N, K);
    __builtin_prefetch(Arow + (kt + 1) * 32, 0, 0);
    v16bf af = load_a_frag_bf16(Arow + kt * 32, g);
#pragma unroll
    for (int t = 0; t < 4; ++t)
      acc[t] = wmma_bf16(af, load_b_frag_bf16(&Bt[cur][0][t * 16 + mrow], g), acc[t]);
    if (wave == 0 && kt + 1 < nk) __builtin_amdgcn_s_wait_tensorcnt(0);
    __syncthreads();
  }
  int mBase = m0 + g * 8;
#pragma unroll
  for (int t = 0; t < 4; ++t) {
#pragma unroll
    for (int r = 0; r < 8; ++r) {
      int nn = n0 + t * 16 + mrow;
      float v = acc[t][r] + bias[nn];
      if (Cb) Cb[(size_t)(mBase + r) * N + nn] = (bf16)v;
      else    Cf[(size_t)(mBase + r) * N + nn] = v;
    }
  }
}

// ---------------- fused gated-FF GEMM: C = (A@B1+b1)*sigmoid(A@B2+b2), bf16 out ----------------
__global__ void __launch_bounds__(128)
gemm_swiglu_wmma(const bf16* __restrict__ A,
                 const bf16* __restrict__ B1, const float* __restrict__ b1,
                 const bf16* __restrict__ B2, const float* __restrict__ b2,
                 bf16* __restrict__ C, int M, int N, int K) {
  __shared__ __align__(16) bf16 Bt1[2][32][64];
  __shared__ __align__(16) bf16 Bt2[2][32][64];
  int wave = threadIdx.x >> 5, lane = threadIdx.x & 31;
  int m0 = blockIdx.x * 64 + wave * 16;
  int n0 = blockIdx.y * 64;
  int mrow = lane & 15, g = lane >> 4;
  v8f acc1[4] = {};
  v8f acc2[4] = {};
  const bf16* Arow = A + (size_t)(m0 + mrow) * K;
  const int nk = K >> 5;
  if (wave == 0) {
    tdm_load_b_tile((unsigned)(size_t)&Bt1[0][0][0], B1 + n0, N, K);
    tdm_load_b_tile((unsigned)(size_t)&Bt2[0][0][0], B2 + n0, N, K);
    __builtin_amdgcn_s_wait_tensorcnt(0);
  }
  __syncthreads();
  for (int kt = 0; kt < nk; ++kt) {
    int cur = kt & 1;
    if (wave == 0 && kt + 1 < nk) {
      size_t goff = (size_t)(kt + 1) * 32 * N + n0;
      tdm_load_b_tile((unsigned)(size_t)&Bt1[cur ^ 1][0][0], B1 + goff, N, K);
      tdm_load_b_tile((unsigned)(size_t)&Bt2[cur ^ 1][0][0], B2 + goff, N, K);
    }
    v16bf af = load_a_frag_bf16(Arow + kt * 32, g);
#pragma unroll
    for (int t = 0; t < 4; ++t) {
      acc1[t] = wmma_bf16(af, load_b_frag_bf16(&Bt1[cur][0][t * 16 + mrow], g), acc1[t]);
      acc2[t] = wmma_bf16(af, load_b_frag_bf16(&Bt2[cur][0][t * 16 + mrow], g), acc2[t]);
    }
    if (wave == 0 && kt + 1 < nk) __builtin_amdgcn_s_wait_tensorcnt(0);
    __syncthreads();
  }
  int mBase = m0 + g * 8;
#pragma unroll
  for (int t = 0; t < 4; ++t) {
#pragma unroll
    for (int r = 0; r < 8; ++r) {
      int nn = n0 + t * 16 + mrow;
      float u  = acc1[t][r] + b1[nn];
      float gg = acc2[t][r] + b2[nn];
      C[(size_t)(mBase + r) * N + nn] = (bf16)(u * (1.0f / (1.0f + __expf(-gg))));
    }
  }
}

// ---------------- fused attention for one (b,h,16-query tile), bf16 q/k/v ----------------
__global__ void __launch_bounds__(128)
attention_kernel(const bf16* __restrict__ Q, const bf16* __restrict__ Kt,
                 const bf16* __restrict__ V, const float* __restrict__ pb,
                 bf16* __restrict__ ctxout) {
  __shared__ float sc[16][Sseq];     // 32 KB
  __shared__ float ctx[16][DKh];     // 4 KB
  __shared__ float red[16][8];
  __shared__ float rowred[16];

  int bh = blockIdx.x;
  int b  = bh >> 3;
  int h  = bh & 7;
  int q0 = blockIdx.y << 4;
  int wave = threadIdx.x >> 5, lane = threadIdx.x & 31;
  int mrow = lane & 15, g = lane >> 4;

  for (int i = threadIdx.x; i < 16 * DKh; i += 128) (&ctx[0][0])[i] = 0.f;

  const bf16* Qr = Q + ((size_t)(b * Sseq + q0 + mrow)) * HD + h * DKh;
  v16bf qf0 = load_a_frag_bf16(Qr, g);
  v16bf qf1 = load_a_frag_bf16(Qr + 32, g);

  // scores: each wave handles keys [wave*128, wave*128+128)
  for (int t = 0; t < 8; ++t) {
    int n0 = wave * 128 + t * 16;
    v8f acc = {};
    {
      v16bf bf;
#pragma unroll
      for (int i = 0; i < 16; ++i)
        bf[i] = Kt[((size_t)(b * Sseq + n0 + mrow)) * HD + h * DKh + (i & 7) + 8 * g + ((i >> 3) << 4)];
      acc = wmma_bf16(qf0, bf, acc);
    }
    {
      v16bf bf;
#pragma unroll
      for (int i = 0; i < 16; ++i)
        bf[i] = Kt[((size_t)(b * Sseq + n0 + mrow)) * HD + h * DKh + 32 + (i & 7) + 8 * g + ((i >> 3) << 4)];
      acc = wmma_bf16(qf1, bf, acc);
    }
#pragma unroll
    for (int r = 0; r < 8; ++r) {
      int m = g * 8 + r;
      sc[m][n0 + mrow] = acc[r] * 0.125f + pb[(size_t)(q0 + m) * Sseq + n0 + mrow];
    }
  }
  __syncthreads();

  // softmax (unnormalized exp; division folded into epilogue)
  int row = threadIdx.x >> 3, part = threadIdx.x & 7;
  int c0 = part * 64;
  float lm = -3.4e38f;
  for (int c = c0; c < c0 + 64; ++c) lm = fmaxf(lm, sc[row][c]);
  red[row][part] = lm;
  __syncthreads();
  if (part == 0) {
    float rm = red[row][0];
#pragma unroll
    for (int i = 1; i < 8; ++i) rm = fmaxf(rm, red[row][i]);
    rowred[row] = rm;
  }
  __syncthreads();
  float rm = rowred[row];
  float ls = 0.f;
  for (int c = c0; c < c0 + 64; ++c) {
    float e = __expf(sc[row][c] - rm);
    sc[row][c] = e;
    ls += e;
  }
  red[row][part] = ls;
  __syncthreads();
  if (part == 0) {
    float s = 0.f;
#pragma unroll
    for (int i = 0; i < 8; ++i) s += red[row][i];
    rowred[row] = s;
  }

  // ctx partial = probs(16 x 128-chunk) @ V(chunk x 64) per wave
  v8f cacc[4] = {};
  for (int kc = 0; kc < 4; ++kc) {
    int k0 = wave * 128 + kc * 32;
    v16bf pf = load_a_frag_f32(&sc[mrow][k0], g);
#pragma unroll
    for (int t = 0; t < 4; ++t) {
      v16bf vf;
#pragma unroll
      for (int i = 0; i < 16; ++i)
        vf[i] = V[((size_t)(b * Sseq + k0 + (i & 7) + 8 * g + ((i >> 3) << 4))) * HD + h * DKh + t * 16 + mrow];
      cacc[t] = wmma_bf16(pf, vf, cacc[t]);
    }
  }
#pragma unroll
  for (int t = 0; t < 4; ++t)
#pragma unroll
    for (int r = 0; r < 8; ++r)
      atomicAdd(&ctx[g * 8 + r][t * 16 + mrow], cacc[t][r]);
  __syncthreads();

  for (int i = threadIdx.x; i < 16 * DKh; i += 128) {
    int m = i >> 6, d = i & 63;
    ctxout[((size_t)(b * Sseq + q0 + m)) * HD + h * DKh + d] =
        (bf16)(ctx[m][d] / rowred[m]);
  }
}

// ---------------- residual (+optional sigmoid gate) + LayerNorm; fp32 + bf16 outputs ----------------
__global__ void __launch_bounds__(256)
resln_kernel(const float* __restrict__ x, const float* __restrict__ a,
             const float* __restrict__ gpre, const float* __restrict__ scale,
             const float* __restrict__ bias, float* __restrict__ y,
             bf16* __restrict__ yb) {
  int row  = blockIdx.x * 8 + (threadIdx.x >> 5);
  int lane = threadIdx.x & 31;
  size_t base = (size_t)row * Dm;
  float v[16];
  float sum = 0.f;
#pragma unroll
  for (int i = 0; i < 16; ++i) {
    int c = lane + (i << 5);
    float add = a[base + c];
    if (gpre) add *= 1.0f / (1.0f + __expf(-gpre[base + c]));
    float t = x[base + c] + add;
    v[i] = t;
    sum += t;
  }
#pragma unroll
  for (int o = 16; o > 0; o >>= 1) sum += __shfl_xor(sum, o, 32);
  float mu = sum * (1.0f / Dm);
  float var = 0.f;
#pragma unroll
  for (int i = 0; i < 16; ++i) { float d = v[i] - mu; var += d * d; }
#pragma unroll
  for (int o = 16; o > 0; o >>= 1) var += __shfl_xor(var, o, 32);
  var *= (1.0f / Dm);
  float rs = rsqrtf(var + EPSf);
#pragma unroll
  for (int i = 0; i < 16; ++i) {
    int c = lane + (i << 5);
    float yv = (v[i] - mu) * rs * scale[c] + bias[c];
    y[base + c]  = yv;
    yb[base + c] = (bf16)yv;
  }
}

// ---------------- mean over S -> hcat[:, 0:512] ----------------
__global__ void pool_kernel(const float* __restrict__ x, float* __restrict__ hcat) {
  int b = blockIdx.x, d = threadIdx.x;
  float s = 0.f;
  for (int t = 0; t < Sseq; ++t) s += x[((size_t)b * Sseq + t) * Dm + d];
  hcat[(size_t)b * (2 * Dm) + d] = s * (1.0f / Sseq);
}

// ---------------- tiny GEMM for the head (M=16) ----------------
__global__ void small_gemm_kernel(const float* __restrict__ A, int lda,
                                  const float* __restrict__ W,
                                  const float* __restrict__ bias,
                                  float* __restrict__ C, int ldc,
                                  int M, int N, int K) {
  int idx = blockIdx.x * blockDim.x + threadIdx.x;
  if (idx >= M * N) return;
  int m = idx / N, n = idx % N;
  const float* Ar = A + (size_t)m * lda;
  float acc = bias[n];
  for (int k = 0; k < K; ++k) acc += Ar[k] * W[(size_t)k * N + n];
  C[(size_t)m * ldc + n] = acc;
}

// ---------------- row LN + relu, in place ----------------
__global__ void __launch_bounds__(128)
lnrelu_kernel(float* __restrict__ h, const float* __restrict__ sc,
              const float* __restrict__ bi, int N) {
  __shared__ float red[128];
  int row = blockIdx.x, tid = threadIdx.x;
  float* hr = h + (size_t)row * N;
  float sum = 0.f;
  for (int c = tid; c < N; c += 128) sum += hr[c];
  red[tid] = sum; __syncthreads();
  for (int o = 64; o > 0; o >>= 1) { if (tid < o) red[tid] += red[tid + o]; __syncthreads(); }
  float mu = red[0] / N;
  __syncthreads();
  float var = 0.f;
  for (int c = tid; c < N; c += 128) { float d = hr[c] - mu; var += d * d; }
  red[tid] = var; __syncthreads();
  for (int o = 64; o > 0; o >>= 1) { if (tid < o) red[tid] += red[tid + o]; __syncthreads(); }
  var = red[0] / N;
  float rs = rsqrtf(var + EPSf);
  for (int c = tid; c < N; c += 128) {
    float y = (hr[c] - mu) * rs * sc[c] + bi[c];
    hr[c] = y > 0.f ? y : 0.f;
  }
}

// ---------------- host-side launcher ----------------
extern "C" void kernel_launch(void* const* d_in, const int* in_sizes, int n_in,
                              void* d_out, int out_size, void* d_ws, size_t ws_size,
                              hipStream_t stream) {
  (void)in_sizes; (void)n_in; (void)out_size; (void)ws_size;
  const float* x_cgm   = (const float*)d_in[0];
  const float* x_other = (const float*)d_in[1];
  const float* cgm_W   = (const float*)d_in[2];
  const float* cgm_b   = (const float*)d_in[3];
  const float* rel_emb = (const float*)d_in[4];
  const float* Wq  = (const float*)d_in[5];  const float* bq  = (const float*)d_in[6];
  const float* Wk  = (const float*)d_in[7];  const float* bk  = (const float*)d_in[8];
  const float* Wv  = (const float*)d_in[9];  const float* bv  = (const float*)d_in[10];
  const float* Wo  = (const float*)d_in[11]; const float* bo  = (const float*)d_in[12];
  const float* Wg  = (const float*)d_in[13]; const float* bg  = (const float*)d_in[14];
  const float* Wf1 = (const float*)d_in[15]; const float* bf1 = (const float*)d_in[16];
  const float* Wfg = (const float*)d_in[17]; const float* bfg = (const float*)d_in[18];
  const float* Wf2 = (const float*)d_in[19]; const float* bf2 = (const float*)d_in[20];
  const float* ln1s = (const float*)d_in[21]; const float* ln1b = (const float*)d_in[22];
  const float* ln2s = (const float*)d_in[23]; const float* ln2b = (const float*)d_in[24];
  const float* oW  = (const float*)d_in[25]; const float* ob  = (const float*)d_in[26];
  const float* fW1 = (const float*)d_in[27]; const float* fb1 = (const float*)d_in[28];
  const float* fl1s = (const float*)d_in[29]; const float* fl1b = (const float*)d_in[30];
  const float* fW2 = (const float*)d_in[31]; const float* fb2 = (const float*)d_in[32];
  const float* fl2s = (const float*)d_in[33]; const float* fl2b = (const float*)d_in[34];
  const float* fW3 = (const float*)d_in[35]; const float* fb3 = (const float*)d_in[36];

  // ---- workspace bump allocator (256B aligned) ----
  char* base = (char*)d_ws;
  size_t off = 0;
  auto alloc = [&](size_t bytes) -> void* {
    void* p = base + off;
    off += (bytes + 255) & ~(size_t)255;
    return p;
  };
  const size_t BSD = (size_t)Bsz * Sseq * Dm;       // 4,194,304
  const size_t BSF = (size_t)Bsz * Sseq * FFd;      // 16,777,216
  const size_t WDD = (size_t)Lnum * Dm * HD;        // 1,048,576
  const size_t WDF = (size_t)Lnum * Dm * FFd;       // 4,194,304

  float* x    = (float*)alloc(BSD * 4);
  float* gb   = (float*)alloc(BSD * 4);
  float* ab   = (float*)alloc(BSD * 4);
  float* pb   = (float*)alloc((size_t)Sseq * Sseq * 4);
  float* hcat = (float*)alloc((size_t)Bsz * 2 * Dm * 4);
  float* h1   = (float*)alloc((size_t)Bsz * 256 * 4);
  float* h2   = (float*)alloc((size_t)Bsz * 128 * 4);
  bf16* xb   = (bf16*)alloc(BSD * 2);
  bf16* qb   = (bf16*)alloc(BSD * 2);
  bf16* kb   = (bf16*)alloc(BSD * 2);
  bf16* vb   = (bf16*)alloc(BSD * 2);
  bf16* cbb  = (bf16*)alloc(BSD * 2);
  bf16* fbb  = (bf16*)alloc(BSF * 2);
  bf16* wqb  = (bf16*)alloc(WDD * 2);
  bf16* wkb  = (bf16*)alloc(WDD * 2);
  bf16* wvb  = (bf16*)alloc(WDD * 2);
  bf16* wgb  = (bf16*)alloc(WDD * 2);
  bf16* wob  = (bf16*)alloc(WDD * 2);
  bf16* wf1b = (bf16*)alloc(WDF * 2);
  bf16* wfgb = (bf16*)alloc(WDF * 2);
  bf16* wf2b = (bf16*)alloc(WDF * 2);

  // ---- one-time (per launch) weight downconversion to bf16 ----
  cvt_bf16_kernel<<<(int)(WDD / 256), 256, 0, stream>>>(Wq,  wqb,  (int)WDD);
  cvt_bf16_kernel<<<(int)(WDD / 256), 256, 0, stream>>>(Wk,  wkb,  (int)WDD);
  cvt_bf16_kernel<<<(int)(WDD / 256), 256, 0, stream>>>(Wv,  wvb,  (int)WDD);
  cvt_bf16_kernel<<<(int)(WDD / 256), 256, 0, stream>>>(Wg,  wgb,  (int)WDD);
  cvt_bf16_kernel<<<(int)(WDD / 256), 256, 0, stream>>>(Wo,  wob,  (int)WDD);
  cvt_bf16_kernel<<<(int)(WDF / 256), 256, 0, stream>>>(Wf1, wf1b, (int)WDF);
  cvt_bf16_kernel<<<(int)(WDF / 256), 256, 0, stream>>>(Wfg, wfgb, (int)WDF);
  cvt_bf16_kernel<<<(int)(WDF / 256), 256, 0, stream>>>(Wf2, wf2b, (int)WDF);

  const int MT = Bsz * Sseq;                      // 8192 rows
  embed_kernel<<<MT * Dm / 256, 256, 0, stream>>>(x_cgm, cgm_W, cgm_b, x, xb);
  posbias_kernel<<<Sseq * Sseq / 256, 256, 0, stream>>>(rel_emb, pb);

  dim3 gD(MT / 64, Dm / 64);    // (128, 8)
  dim3 gF(MT / 64, FFd / 64);   // (128, 32)
  dim3 gA(Bsz * Hh, Sseq / 16); // (128, 32)

  for (int l = 0; l < Lnum; ++l) {
    const size_t oDD = (size_t)l * Dm * HD;
    const size_t oDF = (size_t)l * Dm * FFd;
    // q, k, v, gate projections (q/k/v emit bf16 for attention)
    gemm_wmma_bf16<<<gD, 128, 0, stream>>>(xb, wqb + oDD, bq + (size_t)l * Dm,
                                           nullptr, qb, MT, Dm, Dm);
    gemm_wmma_bf16<<<gD, 128, 0, stream>>>(xb, wkb + oDD, bk + (size_t)l * Dm,
                                           nullptr, kb, MT, Dm, Dm);
    gemm_wmma_bf16<<<gD, 128, 0, stream>>>(xb, wvb + oDD, bv + (size_t)l * Dm,
                                           nullptr, vb, MT, Dm, Dm);
    gemm_wmma_bf16<<<gD, 128, 0, stream>>>(xb, wgb + oDD, bg + (size_t)l * Dm,
                                           gb, nullptr, MT, Dm, Dm);
    attention_kernel<<<gA, 128, 0, stream>>>(qb, kb, vb, pb, cbb);
    gemm_wmma_bf16<<<gD, 128, 0, stream>>>(cbb, wob + oDD, bo + (size_t)l * Dm,
                                           ab, nullptr, MT, Dm, Dm);
    resln_kernel<<<MT / 8, 256, 0, stream>>>(x, ab, gb,
                                             ln1s + (size_t)l * Dm, ln1b + (size_t)l * Dm,
                                             x, xb);
    gemm_swiglu_wmma<<<gF, 128, 0, stream>>>(xb, wf1b + oDF, bf1 + (size_t)l * FFd,
                                             wfgb + oDF, bfg + (size_t)l * FFd,
                                             fbb, MT, FFd, Dm);
    gemm_wmma_bf16<<<gD, 128, 0, stream>>>(fbb, wf2b + oDF, bf2 + (size_t)l * Dm,
                                           ab, nullptr, MT, Dm, FFd);
    resln_kernel<<<MT / 8, 256, 0, stream>>>(x, ab, nullptr,
                                             ln2s + (size_t)l * Dm, ln2b + (size_t)l * Dm,
                                             x, xb);
  }

  pool_kernel<<<Bsz, Dm, 0, stream>>>(x, hcat);
  small_gemm_kernel<<<(Bsz * Dm + 255) / 256, 256, 0, stream>>>(
      x_other, FO, oW, ob, hcat + Dm, 2 * Dm, Bsz, Dm, FO);
  small_gemm_kernel<<<(Bsz * 256 + 255) / 256, 256, 0, stream>>>(
      hcat, 2 * Dm, fW1, fb1, h1, 256, Bsz, 256, 2 * Dm);
  lnrelu_kernel<<<Bsz, 128, 0, stream>>>(h1, fl1s, fl1b, 256);
  small_gemm_kernel<<<(Bsz * 128 + 255) / 256, 256, 0, stream>>>(
      h1, 256, fW2, fb2, h2, 128, Bsz, 128, 256);
  lnrelu_kernel<<<Bsz, 128, 0, stream>>>(h2, fl2s, fl2b, 128);
  small_gemm_kernel<<<1, 256, 0, stream>>>(
      h2, 128, fW3, fb3, (float*)d_out, 1, Bsz, 1, 128);
}
